// DotAttention_57372173140044
// MI455X (gfx1250) — matrix operand order repaired and checked by reading, not compiled, not run on hardware
//
#include <hip/hip_runtime.h>

// ---------------------------------------------------------------------------
// DotAttention for MI455X (gfx1250, wave32, WMMA).
// q = x @ Wq^T ; attn = softmax((q q^T) * sqrt(D)) ; out = attn @ q
// B=4, S=2048, D=1024. ~86 GFLOP vs ~70MB min HBM traffic -> matrix-core
// bound. All three GEMMs run on v_wmma_f32_16x16x32_bf16. Scores stay in
// 128 KB of LDS (flash-style). q is stored both row-major (Q/K operand) and
// transposed (V operand) so every WMMA fragment load is a contiguous b128.
// ---------------------------------------------------------------------------

#define DIMD 1024
#define SEQ  2048
#define BATCH 4

typedef __attribute__((ext_vector_type(16))) __bf16 v16bf;
typedef __attribute__((ext_vector_type(8)))  float  v8f;

union Frag {
  v16bf v;
  unsigned short u[16];
  uint4 q4[2];
};

__device__ __forceinline__ unsigned short f32_to_bf16(float f) {
  union { float f; unsigned int u; } c;
  c.f = f;
  unsigned int r = c.u + 0x7FFFu + ((c.u >> 16) & 1u);  // round-to-nearest-even
  return (unsigned short)(r >> 16);
}

// ---------------------------------------------------------------------------
// Kernel 1: cast f32 inputs to bf16 (bandwidth-trivial prep).
// ---------------------------------------------------------------------------
__global__ void cast_bf16_kernel(const float* __restrict__ x,
                                 const float* __restrict__ w,
                                 unsigned short* __restrict__ xb,
                                 unsigned short* __restrict__ wb,
                                 int nx, int nw) {
  int i = blockIdx.x * blockDim.x + threadIdx.x;
  int stride = gridDim.x * blockDim.x;
  for (int t = i; t < nx; t += stride) xb[t] = f32_to_bf16(x[t]);
  for (int t = i; t < nw; t += stride) wb[t] = f32_to_bf16(w[t]);
}

// ---------------------------------------------------------------------------
// Kernel 2: q = x @ Wq^T  (M=8192, N=1024, K=1024), bf16 WMMA.
// One 16x16 tile per wave; 8 waves/block cover a 16x128 output strip.
// Emits BOTH q (row-major, [8192][1024]) and qT (per-batch [1024][2048]):
// the D-fragment holds 8 consecutive rows per lane at fixed column, so the
// transposed copy is one packed global_store_b128 per lane.
// ---------------------------------------------------------------------------
__global__ void __launch_bounds__(256)
gemm_q_kernel(const unsigned short* __restrict__ xb,
              const unsigned short* __restrict__ wb,
              unsigned short* __restrict__ qb,
              unsigned short* __restrict__ qbT) {
  const int lane = threadIdx.x & 31;
  const int wv   = threadIdx.x >> 5;
  const int half = (lane >> 4) & 1;
  const int l15  = lane & 15;
  const int m0 = blockIdx.x * 16;
  const int n0 = blockIdx.y * 128 + wv * 16;

  v8f acc = {};
  // A 16x32 bf16, contiguous k order used here (B matches): lane half 0 -> K 0..15; half 1 -> K 16..31
  const unsigned short* arow = xb + (size_t)(m0 + l15) * DIMD + half * 16;
  // B 32x16, same contiguous k order as A: lane = column n, half-wave selects K 0..15 / 16..31
  const unsigned short* brow = wb + (size_t)(n0 + l15) * DIMD + half * 16;

  for (int kb = 0; kb < DIMD; kb += 32) {
    Frag a, b;
    a.q4[0] = *(const uint4*)(arow + kb);
    a.q4[1] = *(const uint4*)(arow + kb + 8);
    b.q4[0] = *(const uint4*)(brow + kb);
    b.q4[1] = *(const uint4*)(brow + kb + 8);
    acc = __builtin_amdgcn_wmma_f32_16x16x32_bf16(
        false, a.v, false, b.v, (short)0, acc, false, false);
  }

  // Row-major store (strided b16 per row).
  const int rbase = m0 + (half ? 8 : 0);
#pragma unroll
  for (int r = 0; r < 8; ++r)
    qb[(size_t)(rbase + r) * DIMD + n0 + l15] = f32_to_bf16(acc[r]);

  // Transposed store: 8 consecutive bf16 along the row dim -> one b128.
  const int bq   = m0 / SEQ;        // batch
  const int mloc = m0 % SEQ;        // row within batch
  union { uint4 q; unsigned short u[8]; } pk;
#pragma unroll
  for (int r = 0; r < 8; ++r) pk.u[r] = f32_to_bf16(acc[r]);
  *(uint4*)(qbT + ((size_t)bq * DIMD + n0 + l15) * SEQ + mloc + half * 8) = pk.q;
}

// ---------------------------------------------------------------------------
// Kernel 3: fused attention. One block = one 16-row query tile of one batch.
// Phase 1: S[16][2048] = (Q K^T)*sqrt(D) in LDS (128 KB of the 320 KB WGP LDS).
// Phase 2: wave-parallel row softmax; probabilities written back IN PLACE as
//          packed bf16 (exp values held in registers so all f32 reads finish
//          before the bf16 overwrite).
// Phase 3: out = P @ V with WMMA; A-fragments are two ds_load_b128 from the
//          bf16 probability rows, B-fragments are two global b128 from qT.
// K = V = q rows of this batch (4 MB -> fully resident in 192 MB L2).
// ---------------------------------------------------------------------------
__global__ void __launch_bounds__(256)
attn_kernel(const unsigned short* __restrict__ qb,
            const unsigned short* __restrict__ qbT,
            float* __restrict__ out) {
  extern __shared__ float S[];  // 16 * SEQ floats = 128 KB dynamic LDS

  const int lane = threadIdx.x & 31;
  const int wv   = threadIdx.x >> 5;
  const int half = (lane >> 4) & 1;
  const int l15  = lane & 15;
  const int b    = blockIdx.y;
  const int mrow = blockIdx.x * 16;  // query row base within batch
  const unsigned short* qbatch  = qb  + (size_t)b * SEQ * DIMD;
  const unsigned short* qTbatch = qbT + (size_t)b * DIMD * SEQ;

  // ---- Phase 1: scores -------------------------------------------------
  const unsigned short* arow = qbatch + (size_t)(mrow + l15) * DIMD + half * 16;
  for (int pass = 0; pass < 2; ++pass) {
    v8f zero = {};
    v8f acc[8];
#pragma unroll
    for (int t = 0; t < 8; ++t) acc[t] = zero;

    for (int kb = 0; kb < DIMD; kb += 32) {
      Frag a;
      a.q4[0] = *(const uint4*)(arow + kb);
      a.q4[1] = *(const uint4*)(arow + kb + 8);
#pragma unroll
      for (int t = 0; t < 8; ++t) {
        const int nt = pass * 64 + t * 8 + wv;  // key tile index 0..127
        const unsigned short* brow =
            qbatch + (size_t)(nt * 16 + l15) * DIMD + half * 16 + kb;
        __builtin_prefetch(brow + 32, 0, 3);  // next K chunk -> global_prefetch
        Frag bf;
        bf.q4[0] = *(const uint4*)(brow);
        bf.q4[1] = *(const uint4*)(brow + 8);
        acc[t] = __builtin_amdgcn_wmma_f32_16x16x32_bf16(
            false, a.v, false, bf.v, (short)0, acc[t], false, false);
      }
    }
#pragma unroll
    for (int t = 0; t < 8; ++t) {
      const int n0 = (pass * 64 + t * 8 + wv) * 16;
#pragma unroll
      for (int r = 0; r < 8; ++r)
        S[((half ? 8 : 0) + r) * SEQ + n0 + l15] = acc[t][r] * 32.0f;  // *sqrt(D)
    }
  }
  __syncthreads();

  // ---- Phase 2: softmax (wave wv owns rows 2wv, 2wv+1) -----------------
  for (int rr = 0; rr < 2; ++rr) {
    float* row = S + (wv * 2 + rr) * SEQ;
    float mx = -3.402823466e38f;
#pragma unroll
    for (int j = 0; j < SEQ / 32; ++j) mx = fmaxf(mx, row[lane + 32 * j]);
#pragma unroll
    for (int off = 16; off > 0; off >>= 1) mx = fmaxf(mx, __shfl_xor(mx, off, 32));

    float ev[SEQ / 32];  // kept in registers: all f32 reads precede writes
    float sum = 0.0f;
#pragma unroll
    for (int j = 0; j < SEQ / 32; ++j) {
      ev[j] = __expf(row[lane + 32 * j] - mx);
      sum += ev[j];
    }
#pragma unroll
    for (int off = 16; off > 0; off >>= 1) sum += __shfl_xor(sum, off, 32);
    const float inv = 1.0f / sum;

    // In-place repack: f32 scores -> bf16 probabilities (same LDS row).
    unsigned short* prow = (unsigned short*)row;
#pragma unroll
    for (int j = 0; j < SEQ / 32; ++j)
      prow[lane + 32 * j] = f32_to_bf16(ev[j] * inv);
  }
  __syncthreads();

  // ---- Phase 3: out = P @ V  (wave wv owns output cols [wv*128, wv*128+128))
  v8f zero = {};
  v8f acc2[8];
#pragma unroll
  for (int t = 0; t < 8; ++t) acc2[t] = zero;

  const unsigned short* prow = (const unsigned short*)(S + (size_t)l15 * SEQ);
  for (int kb = 0; kb < SEQ; kb += 32) {
    Frag a;  // A = P tile: two contiguous 8-element bf16 runs -> ds_load_b128
    a.q4[0] = *(const uint4*)(prow + kb + half * 16);
    a.q4[1] = *(const uint4*)(prow + kb + half * 16 + 8);
#pragma unroll
    for (int t = 0; t < 8; ++t) {
      const int n = wv * 128 + t * 16 + l15;
      // B = V tile via transposed copy: contiguous along K -> b128 loads.
      const unsigned short* brow = qTbatch + (size_t)n * SEQ + half * 16 + kb;
      Frag bf;
      bf.q4[0] = *(const uint4*)(brow);
      bf.q4[1] = *(const uint4*)(brow + 8);
      acc2[t] = __builtin_amdgcn_wmma_f32_16x16x32_bf16(
          false, a.v, false, bf.v, (short)0, acc2[t], false, false);
    }
  }

  float* orow = out + ((size_t)b * SEQ + mrow) * DIMD;
#pragma unroll
  for (int t = 0; t < 8; ++t) {
    const int n = wv * 128 + t * 16 + l15;
#pragma unroll
    for (int r = 0; r < 8; ++r)
      orow[(size_t)((half ? 8 : 0) + r) * DIMD + n] = acc2[t][r];
  }
}

// ---------------------------------------------------------------------------
extern "C" void kernel_launch(void* const* d_in, const int* in_sizes, int n_in,
                              void* d_out, int out_size, void* d_ws, size_t ws_size,
                              hipStream_t stream) {
  const float* x  = (const float*)d_in[0];   // [4, 2048, 1024] f32
  const float* Wq = (const float*)d_in[1];   // [1024, 1024] f32
  float* out = (float*)d_out;                // [4, 2048, 1024] f32

  // Workspace layout (50 MB total):
  unsigned short* xb  = (unsigned short*)d_ws;                      // 16 MB
  unsigned short* wb  = xb  + (size_t)BATCH * SEQ * DIMD;           //  2 MB
  unsigned short* qbm = wb  + (size_t)DIMD * DIMD;                  // 16 MB row-major q
  unsigned short* qbT = qbm + (size_t)BATCH * SEQ * DIMD;           // 16 MB transposed q

  const int nx = BATCH * SEQ * DIMD;
  const int nw = DIMD * DIMD;

  cast_bf16_kernel<<<2048, 256, 0, stream>>>(x, Wq, xb, wb, nx, nw);

  gemm_q_kernel<<<dim3((BATCH * SEQ) / 16, DIMD / 128), 256, 0, stream>>>(xb, wb, qbm, qbT);

  attn_kernel<<<dim3(SEQ / 16, BATCH), 256, 16 * SEQ * sizeof(float), stream>>>(qbm, qbT, out);
}
